// FlaxDeepseekV2DecoderLayer_76879914598680
// MI455X (gfx1250) — matrix operand order repaired and checked by reading, not compiled, not run on hardware
//
#include <hip/hip_runtime.h>

typedef unsigned short u16;
typedef unsigned int   u32;

typedef __attribute__((ext_vector_type(16))) __bf16 v16bf;
typedef __attribute__((ext_vector_type(2)))  __bf16 v2bf;
typedef __attribute__((ext_vector_type(8)))  float  v8f;

union Frag { v16bf v; u32 u[8]; };

#define SEQ   2048
#define HDIM  2048
#define NHEAD 16
#define QHD_  192
#define VHD_  128

__device__ __forceinline__ u16 f2bf(float f) {
  u32 u = __float_as_uint(f);
  u += 0x7FFFu + ((u >> 16) & 1u);   // round-to-nearest-even
  return (u16)(u >> 16);
}
// packed f32x2 -> bf16x2 (single v_cvt_pk_bf16_f32 when available)
__device__ __forceinline__ u32 pack2(float lo, float hi) {
#if __has_builtin(__builtin_amdgcn_cvt_pk_bf16_f32)
  union { v2bf v; u32 u; } c;
  c.v = __builtin_amdgcn_cvt_pk_bf16_f32(lo, hi);
  return c.u;
#else
  return (u32)f2bf(lo) | ((u32)f2bf(hi) << 16);
#endif
}
__device__ __forceinline__ uint4 pack4(float4 lo, float4 hi) {
  return make_uint4(pack2(lo.x, hi.x), pack2(lo.y, hi.y),
                    pack2(lo.z, hi.z), pack2(lo.w, hi.w));
}

// CDNA5 async global->LDS copy (ASYNCcnt tracked), 16B per lane.
__device__ __forceinline__ void async_b128(u32 lds_addr, const void* gaddr) {
  asm volatile("global_load_async_to_lds_b128 %0, %1, off"
               :: "v"(lds_addr), "v"(gaddr) : "memory");
}
__device__ __forceinline__ void wait_async0() {
  asm volatile("s_wait_asynccnt 0x0" ::: "memory");
}

// ---------------------------------------------------------------------------
// RMSNorm: one block per row, output bf16 (paired stores)
// ---------------------------------------------------------------------------
__global__ void __launch_bounds__(256)
rmsnorm_bf16_kernel(const float* __restrict__ x, const float* __restrict__ w,
                    u16* __restrict__ y, int n, int in_stride, int out_stride)
{
  __shared__ float red[8];
  const int row = blockIdx.x;
  const float* xr = x + (size_t)row * in_stride;
  float ss = 0.f;
  for (int i = threadIdx.x; i < n; i += 256) { float v = xr[i]; ss += v * v; }
  #pragma unroll
  for (int off = 16; off; off >>= 1) ss += __shfl_xor(ss, off, 32);
  if ((threadIdx.x & 31) == 0) red[threadIdx.x >> 5] = ss;
  __syncthreads();
  float tot = 0.f;
  #pragma unroll
  for (int i = 0; i < 8; i++) tot += red[i];
  const float scale = rsqrtf(tot / (float)n + 1e-6f);
  u32* yr = (u32*)(y + (size_t)row * out_stride);
  const int n2 = n >> 1;
  for (int i = threadIdx.x; i < n2; i += 256)
    yr[i] = pack2(xr[2 * i] * scale * w[2 * i],
                  xr[2 * i + 1] * scale * w[2 * i + 1]);
}

// ---------------------------------------------------------------------------
// FAST GEMM (N%128==0, M%128==0, K%32==0):
// C[M,N] f32 = A[M,K] bf16 x B[K,N] f32 (bf16-converted during staging).
// Block tile 128x128, BK=32, 8 waves, each wave 2x4 WMMA tiles (32x64).
// Double-buffered LDS; A staged via global_load_async_to_lds_b128; B
// prefetched to registers before compute, converted+stored after compute.
// ---------------------------------------------------------------------------
__global__ void __launch_bounds__(256)
gemm_bf16_fast(const u16* __restrict__ A, const float* __restrict__ B,
               float* __restrict__ C, int M, int N, int K)
{
  __shared__ u32 As[2][128][16];   // row-major: dword = A[m][2k],A[m][2k+1]
  __shared__ u32 Bs[2][16][128];   // pair-packed: dword = B[2k][n],B[2k+1][n]

  const int tid  = threadIdx.x;
  const int lane = tid & 31, wave = tid >> 5;
  const int half = lane >> 4, l16 = lane & 15;
  const int bm0 = blockIdx.y * 128;
  const int bn0 = blockIdx.x * 128;
  const int wm = (wave >> 1) * 32;      // 0 / 32 / 64 / 96
  const int wn = (wave & 1) * 64;       // 0 / 64

  // A staging: 512 x 16B chunks -> 2 async per thread per tile
  const int ar0 = tid >> 2;             // rows 0..63
  const int ar1 = ar0 + 64;             // rows 64..127
  const int ap  = tid & 3;              // which 16B of the 64B row
  // B staging: two 4-column quads per thread
  const int q0r = tid >> 5;             // pair-row 0..7
  const int q1r = q0r + 8;              // pair-row 8..15
  const int qc  = (tid & 31) * 4;       // column quad

  v8f acc[2][4] = {};
  const int ntiles = K >> 5;

  // ---- prologue: stage tile 0 ----
  {
    async_b128((u32)(size_t)&As[0][ar0][ap * 4],
               A + (size_t)(bm0 + ar0) * K + ap * 8);
    async_b128((u32)(size_t)&As[0][ar1][ap * 4],
               A + (size_t)(bm0 + ar1) * K + ap * 8);
    float4 l0 = *(const float4*)(B + (size_t)(2 * q0r)     * N + bn0 + qc);
    float4 h0 = *(const float4*)(B + (size_t)(2 * q0r + 1) * N + bn0 + qc);
    float4 l1 = *(const float4*)(B + (size_t)(2 * q1r)     * N + bn0 + qc);
    float4 h1 = *(const float4*)(B + (size_t)(2 * q1r + 1) * N + bn0 + qc);
    *(uint4*)&Bs[0][q0r][qc] = pack4(l0, h0);
    *(uint4*)&Bs[0][q1r][qc] = pack4(l1, h1);
  }
  wait_async0();
  __syncthreads();

  for (int t = 0; t < ntiles; t++) {
    const int buf = t & 1;
    const bool pre = (t + 1 < ntiles);
    float4 l0, h0, l1, h1;
    if (pre) {
      const int k0n = (t + 1) * 32;
      async_b128((u32)(size_t)&As[buf ^ 1][ar0][ap * 4],
                 A + (size_t)(bm0 + ar0) * K + k0n + ap * 8);
      async_b128((u32)(size_t)&As[buf ^ 1][ar1][ap * 4],
                 A + (size_t)(bm0 + ar1) * K + k0n + ap * 8);
      l0 = *(const float4*)(B + (size_t)(k0n + 2 * q0r)     * N + bn0 + qc);
      h0 = *(const float4*)(B + (size_t)(k0n + 2 * q0r + 1) * N + bn0 + qc);
      l1 = *(const float4*)(B + (size_t)(k0n + 2 * q1r)     * N + bn0 + qc);
      h1 = *(const float4*)(B + (size_t)(k0n + 2 * q1r + 1) * N + bn0 + qc);
    }
    // ---- compute current tile: hoist 4 B fragments, reuse for both M tiles
    Frag bfr[4];
    #pragma unroll
    for (int ni = 0; ni < 4; ni++) {
      const int bcol = wn + ni * 16 + l16;
      #pragma unroll
      for (int i = 0; i < 8; i++)
        bfr[ni].u[i] = Bs[buf][half * 8 + i][bcol];
    }
    #pragma unroll
    for (int mi = 0; mi < 2; mi++) {
      Frag a;
      const int arow = wm + mi * 16 + l16;
      #pragma unroll
      for (int i = 0; i < 8; i++) {
        int kd = half * 8 + i;
        a.u[i] = As[buf][arow][kd];
      }
      #pragma unroll
      for (int ni = 0; ni < 4; ni++)
        acc[mi][ni] = __builtin_amdgcn_wmma_f32_16x16x32_bf16(
            false, a.v, false, bfr[ni].v, (short)0, acc[mi][ni], false, false);
    }
    if (pre) {
      *(uint4*)&Bs[buf ^ 1][q0r][qc] = pack4(l0, h0);
      *(uint4*)&Bs[buf ^ 1][q1r][qc] = pack4(l1, h1);
    }
    wait_async0();
    __syncthreads();
  }

  #pragma unroll
  for (int mi = 0; mi < 2; mi++)
    #pragma unroll
    for (int ni = 0; ni < 4; ni++) {
      int col = bn0 + wn + ni * 16 + l16;
      #pragma unroll
      for (int r = 0; r < 8; r++) {
        int row = bm0 + wm + mi * 16 + half * 8 + r;
        C[(size_t)row * N + col] = acc[mi][ni][r];
      }
    }
}

// ---------------------------------------------------------------------------
// Generic GEMM (handles N not multiple of 128; used only for N=576).
// ---------------------------------------------------------------------------
__global__ void __launch_bounds__(256)
gemm_bf16_gen(const u16* __restrict__ A, const float* __restrict__ B,
              float* __restrict__ C, int M, int N, int K)
{
  __shared__ u32 As[64][16];
  __shared__ u32 Bs[16][128];

  const int tid  = threadIdx.x;
  const int lane = tid & 31, wave = tid >> 5;
  const int half = lane >> 4, l16 = lane & 15;
  const int bm0 = blockIdx.y * 64;
  const int bn0 = blockIdx.x * 128;
  const int wm = (wave >> 2) * 32;
  const int wn = (wave & 3) * 32;

  v8f acc[2][2] = {};

  for (int k0 = 0; k0 < K; k0 += 32) {
    #pragma unroll
    for (int j = 0; j < 4; j++) {
      int i = tid + j * 256;
      int r = i >> 4, c = i & 15;
      As[r][c] = *(const u32*)(A + (size_t)(bm0 + r) * K + k0 + 2 * c);
    }
    #pragma unroll
    for (int j = 0; j < 8; j++) {
      int i = tid + j * 256;
      int r = i >> 7, c = i & 127;
      int gn = bn0 + c;
      u32 val = 0;
      if (gn < N) {
        float lo = B[(size_t)(k0 + 2 * r)     * N + gn];
        float hi = B[(size_t)(k0 + 2 * r + 1) * N + gn];
        val = pack2(lo, hi);
      }
      Bs[r][c] = val;
    }
    __syncthreads();

    #pragma unroll
    for (int mi = 0; mi < 2; mi++) {
      Frag a;
      const int arow = wm + mi * 16 + l16;
      #pragma unroll
      for (int i = 0; i < 8; i++) {
        int kd = half * 8 + i;
        a.u[i] = As[arow][kd];
      }
      #pragma unroll
      for (int ni = 0; ni < 2; ni++) {
        Frag b;
        const int bcol = wn + ni * 16 + l16;
        #pragma unroll
        for (int i = 0; i < 8; i++)
          b.u[i] = Bs[half * 8 + i][bcol];
        acc[mi][ni] = __builtin_amdgcn_wmma_f32_16x16x32_bf16(
            false, a.v, false, b.v, (short)0, acc[mi][ni], false, false);
      }
    }
    __syncthreads();
  }

  #pragma unroll
  for (int mi = 0; mi < 2; mi++)
    #pragma unroll
    for (int ni = 0; ni < 2; ni++) {
      int col = bn0 + wn + ni * 16 + l16;
      if (col < N) {
        #pragma unroll
        for (int r = 0; r < 8; r++) {
          int row = bm0 + wm + mi * 16 + half * 8 + r;
          C[(size_t)row * N + col] = acc[mi][ni][r];
        }
      }
    }
}

// ---------------------------------------------------------------------------
// RoPE / layout builders: Q,K as (NH,S,192) bf16; V pair-packed (NH,S/2,128)
// interleave + rotate_half:  out[j] = x[2j]*c - x[2j+1]*s  (j<32)
//                            out[32+j] = x[2j+1]*c + x[2j]*s
// ---------------------------------------------------------------------------
__global__ void build_q_kernel(const float* __restrict__ qf,
                               const int* __restrict__ pos,
                               const float* __restrict__ cosb,
                               const float* __restrict__ sinb,
                               u16* __restrict__ out)
{
  int idx = blockIdx.x * blockDim.x + threadIdx.x;
  const int total = NHEAD * SEQ * QHD_;
  if (idx >= total) return;
  int d = idx % QHD_;
  int s = (idx / QHD_) % SEQ;
  int h = idx / (QHD_ * SEQ);
  const float* row = qf + (size_t)s * (NHEAD * QHD_) + h * QHD_;
  float val;
  if (d < 128) {
    val = row[d];
  } else {
    int j = d - 128;
    int p = (j < 32) ? j : (j - 32);
    int ps = pos[s];
    float c = cosb[ps * 64 + j], sn = sinb[ps * 64 + j];
    float e = row[128 + 2 * p], o = row[128 + 2 * p + 1];
    val = (j < 32) ? (e * c - o * sn) : (o * c + e * sn);
  }
  out[idx] = f2bf(val);
}

__global__ void build_k_kernel(const float* __restrict__ kvf,
                               const float* __restrict__ ckv,
                               const int* __restrict__ pos,
                               const float* __restrict__ cosb,
                               const float* __restrict__ sinb,
                               u16* __restrict__ out)
{
  int idx = blockIdx.x * blockDim.x + threadIdx.x;
  const int total = NHEAD * SEQ * QHD_;
  if (idx >= total) return;
  int d = idx % QHD_;
  int s = (idx / QHD_) % SEQ;
  int h = idx / (QHD_ * SEQ);
  float val;
  if (d < 128) {
    val = kvf[(size_t)s * 4096 + h * 256 + d];
  } else {
    int j = d - 128;
    int p = (j < 32) ? j : (j - 32);
    int ps = pos[s];
    const float* row = ckv + (size_t)s * 576 + 512;   // shared k_pe
    float c = cosb[ps * 64 + j], sn = sinb[ps * 64 + j];
    float e = row[2 * p], o = row[2 * p + 1];
    val = (j < 32) ? (e * c - o * sn) : (o * c + e * sn);
  }
  out[idx] = f2bf(val);
}

// V pair-packed: out[(h*(S/2)+kp)*128+n] = {V[2kp][n], V[2kp+1][n]}
__global__ void build_v_kernel(const float* __restrict__ kvf, u32* __restrict__ out)
{
  int idx = blockIdx.x * blockDim.x + threadIdx.x;
  const int total = NHEAD * (SEQ / 2) * VHD_;
  if (idx >= total) return;
  int n  = idx % VHD_;
  int kp = (idx / VHD_) % (SEQ / 2);
  int h  = idx / (VHD_ * (SEQ / 2));
  float lo = kvf[(size_t)(2 * kp)     * 4096 + h * 256 + 128 + n];
  float hi = kvf[(size_t)(2 * kp + 1) * 4096 + h * 256 + 128 + n];
  out[idx] = pack2(lo, hi);
}

// ---------------------------------------------------------------------------
// Causal flash attention, one wave per (head, 16-row q block).
// Q/K/V staged by async-to-LDS; K/V double-buffered (copy of block t+1
// overlaps compute of block t). QK^T: 2x6 wmma; online softmax; P.V: 8 wmma.
// ---------------------------------------------------------------------------
__global__ void __launch_bounds__(32)
mla_flash_attn(const u16* __restrict__ Qg, const u16* __restrict__ Kg,
               const u32* __restrict__ Vp, u16* __restrict__ Obf)
{
  __shared__ u32 Qs[16][96];       // 16 x 192 bf16 row-major
  __shared__ u32 Ks[2][32][96];    // 32 x 192 bf16 row-major
  __shared__ u32 Vs[2][16][128];   // pair-packed dwords
  __shared__ u16 Ps[16][32];       // probs tile

  const int h  = blockIdx.y;
  const int q0 = blockIdx.x * 16;
  const int lane = threadIdx.x;
  const int half = lane >> 4, l16 = lane & 15;

  const u16* Qh = Qg + ((size_t)h * SEQ + q0) * QHD_;
  const u16* Kh = Kg + (size_t)h * SEQ * QHD_;
  const u32* Vh = Vp + (size_t)h * (SEQ / 2) * VHD_;

  // stage Q: 384 x 16B chunks, 24 per row
  #pragma unroll
  for (int c0 = 0; c0 < 384; c0 += 32) {
    int c = c0 + lane, r = c / 24, p = c % 24;
    async_b128((u32)(size_t)&Qs[r][p * 4], Qh + (size_t)r * QHD_ + p * 8);
  }

  auto stage_kv = [&](int kv0, int b) {
    #pragma unroll
    for (int c0 = 0; c0 < 768; c0 += 32) {        // K: 24 chunks per row x 32
      int c = c0 + lane, r = c / 24, p = c % 24;
      async_b128((u32)(size_t)&Ks[b][r][p * 4],
                 Kh + (size_t)(kv0 + r) * QHD_ + p * 8);
    }
    #pragma unroll
    for (int c0 = 0; c0 < 512; c0 += 32) {        // V: 32 chunks per pair-row x 16
      int c = c0 + lane, r = c >> 5, p = c & 31;
      async_b128((u32)(size_t)&Vs[b][r][p * 4],
                 Vh + (size_t)(kv0 / 2 + r) * VHD_ + p * 4);
    }
  };

  v8f O[8] = {};
  float mrow[8], lrow[8];
  #pragma unroll
  for (int r = 0; r < 8; r++) { mrow[r] = -3.0e38f; lrow[r] = 0.f; }

  const float sc = 0.072168783648703220563f;   // 192^-0.5
  const int iters = (q0 + 16 + 31) / 32;

  stage_kv(0, 0);

  for (int t = 0; t < iters; t++) {
    const int kv0 = t * 32;
    const int buf = t & 1;
    wait_async0();                       // Q + K/V[buf] resident
    if (t + 1 < iters) stage_kv(kv0 + 32, buf ^ 1);

    float p0[8], p1[8];
    #pragma unroll
    for (int nt = 0; nt < 2; nt++) {
      v8f s8 = {};
      #pragma unroll
      for (int ks = 0; ks < 6; ks++) {
        Frag a, b;
        #pragma unroll
        for (int i = 0; i < 8; i++) {
          int kd = ks * 16 + half * 8 + i;
          a.u[i] = Qs[l16][kd];
          b.u[i] = Ks[buf][nt * 16 + l16][ks * 16 + half * 8 + i];
        }
        s8 = __builtin_amdgcn_wmma_f32_16x16x32_bf16(
            false, a.v, false, b.v, (short)0, s8, false, false);
      }
      float* dst = nt ? p1 : p0;
      const int kvcol = kv0 + nt * 16 + l16;
      #pragma unroll
      for (int r = 0; r < 8; r++) {
        int qrow = q0 + half * 8 + r;
        dst[r] = (kvcol <= qrow) ? s8[r] * sc : -3.0e38f;
      }
    }

    float fac[8];
    #pragma unroll
    for (int r = 0; r < 8; r++) {
      float mx = fmaxf(p0[r], p1[r]);
      #pragma unroll
      for (int off = 8; off; off >>= 1) mx = fmaxf(mx, __shfl_xor(mx, off, 32));
      float mn = fmaxf(mrow[r], mx);
      fac[r] = __expf(mrow[r] - mn);
      mrow[r] = mn;
      p0[r] = __expf(p0[r] - mn);
      p1[r] = __expf(p1[r] - mn);
      float rs = p0[r] + p1[r];
      #pragma unroll
      for (int off = 8; off; off >>= 1) rs += __shfl_xor(rs, off, 32);
      lrow[r] = lrow[r] * fac[r] + rs;
    }
    #pragma unroll
    for (int nv = 0; nv < 8; nv++)
      #pragma unroll
      for (int r = 0; r < 8; r++)
        O[nv][r] *= fac[r];

    #pragma unroll
    for (int r = 0; r < 8; r++) {
      Ps[half * 8 + r][l16]      = f2bf(p0[r]);
      Ps[half * 8 + r][16 + l16] = f2bf(p1[r]);
    }

    Frag pa;
    #pragma unroll
    for (int i = 0; i < 8; i++) {
      int kd = half * 8 + i;
      pa.u[i] = *(const u32*)&Ps[l16][kd * 2];
    }
    #pragma unroll
    for (int nv = 0; nv < 8; nv++) {
      Frag vb;
      #pragma unroll
      for (int i = 0; i < 8; i++) vb.u[i] = Vs[buf][half * 8 + i][nv * 16 + l16];
      O[nv] = __builtin_amdgcn_wmma_f32_16x16x32_bf16(
          false, pa.v, false, vb.v, (short)0, O[nv], false, false);
    }
  }

  #pragma unroll
  for (int r = 0; r < 8; r++) lrow[r] = 1.f / lrow[r];
  #pragma unroll
  for (int nv = 0; nv < 8; nv++)
    #pragma unroll
    for (int r = 0; r < 8; r++) {
      int qrow = q0 + half * 8 + r;
      Obf[(size_t)qrow * (NHEAD * VHD_) + h * VHD_ + nv * 16 + l16] =
          f2bf(O[nv][r] * lrow[r]);
    }
}

// ---------------------------------------------------------------------------
// Elementwise helpers (vectorized)
// ---------------------------------------------------------------------------
__global__ void add_inplace_kernel(float4* __restrict__ a, const float4* __restrict__ b, int n4)
{
  int i = blockIdx.x * blockDim.x + threadIdx.x;
  if (i < n4) {
    float4 x = a[i], y = b[i];
    x.x += y.x; x.y += y.y; x.z += y.z; x.w += y.w;
    a[i] = x;
  }
}

__global__ void silu_mul_kernel(const float2* __restrict__ g, const float2* __restrict__ u,
                                u32* __restrict__ out, int n2)
{
  int i = blockIdx.x * blockDim.x + threadIdx.x;
  if (i < n2) {
    float2 gv = g[i], uv = u[i];
    float s0 = gv.x / (1.f + __expf(-gv.x)) * uv.x;
    float s1 = gv.y / (1.f + __expf(-gv.y)) * uv.y;
    out[i] = pack2(s0, s1);
  }
}

// ---------------------------------------------------------------------------
// Workspace layout (bytes). Phase-B buffers alias dead phase-A buffers.
// ---------------------------------------------------------------------------
static constexpr size_t OFF_XBF  = 0;                                    // S*2048 bf16
static constexpr size_t OFF_QA   = OFF_XBF  + (size_t)SEQ * 2048 * 2;    // S*1536 f32
static constexpr size_t OFF_QIN  = OFF_QA   + (size_t)SEQ * 1536 * 4;    // S*1536 bf16
static constexpr size_t OFF_QF   = OFF_QIN  + (size_t)SEQ * 1536 * 2;    // S*3072 f32
static constexpr size_t OFF_CKV  = OFF_QF   + (size_t)SEQ * 3072 * 4;    // S*576  f32
static constexpr size_t OFF_KVIN = OFF_CKV  + (size_t)SEQ * 576  * 4;    // S*512  bf16
static constexpr size_t OFF_KVF  = OFF_KVIN + (size_t)SEQ * 512  * 2;    // S*4096 f32
static constexpr size_t OFF_QBF  = OFF_KVF  + (size_t)SEQ * 4096 * 4;    // NH*S*192 bf16
static constexpr size_t OFF_KBF  = OFF_QBF  + (size_t)NHEAD * SEQ * 192 * 2;
static constexpr size_t OFF_VBF  = OFF_KBF  + (size_t)NHEAD * SEQ * 192 * 2;
static constexpr size_t OFF_ATTN = OFF_VBF  + (size_t)NHEAD * (SEQ / 2) * 128 * 4;
static constexpr size_t END_A    = OFF_ATTN + (size_t)SEQ * 2048 * 2;
// Phase B (after attention + o-proj): alias phase-A region
static constexpr size_t OFF_GATE = 0;                                    // S*8192 f32
static constexpr size_t OFF_UP   = OFF_GATE + (size_t)SEQ * 8192 * 4;    // S*8192 f32
static constexpr size_t OFF_ACT  = END_A;                                // S*8192 bf16
static constexpr size_t END_B    = OFF_ACT + (size_t)SEQ * 8192 * 2;
static constexpr size_t OFF_X2   = (END_B > END_A ? END_B : END_A);      // S*2048 f32
static constexpr size_t OFF_HBF  = OFF_X2 + (size_t)SEQ * 2048 * 4;      // S*2048 bf16

extern "C" void kernel_launch(void* const* d_in, const int* in_sizes, int n_in,
                              void* d_out, int out_size, void* d_ws, size_t ws_size,
                              hipStream_t stream) {
  (void)in_sizes; (void)n_in; (void)out_size; (void)ws_size;

  const float* hidden    = (const float*)d_in[0];
  const int*   pos       = (const int*)  d_in[1];
  const float* cosb      = (const float*)d_in[2];
  const float* sinb      = (const float*)d_in[3];
  const float* input_ln  = (const float*)d_in[4];
  const float* q_a_w     = (const float*)d_in[5];
  const float* q_a_ln    = (const float*)d_in[6];
  const float* q_b_w     = (const float*)d_in[7];
  const float* kv_a_w    = (const float*)d_in[8];
  const float* kv_a_ln   = (const float*)d_in[9];
  const float* kv_b_w    = (const float*)d_in[10];
  const float* o_w       = (const float*)d_in[11];
  const float* post_ln   = (const float*)d_in[12];
  const float* gate_w    = (const float*)d_in[13];
  const float* up_w      = (const float*)d_in[14];
  const float* down_w    = (const float*)d_in[15];
  float* out = (float*)d_out;

  char* ws = (char*)d_ws;
  u16*   x_bf  = (u16*)  (ws + OFF_XBF);
  float* qa    = (float*)(ws + OFF_QA);
  u16*   qin   = (u16*)  (ws + OFF_QIN);
  float* qf    = (float*)(ws + OFF_QF);
  float* ckv   = (float*)(ws + OFF_CKV);
  u16*   kvin  = (u16*)  (ws + OFF_KVIN);
  float* kvf   = (float*)(ws + OFF_KVF);
  u16*   Qbf   = (u16*)  (ws + OFF_QBF);
  u16*   Kbf   = (u16*)  (ws + OFF_KBF);
  u32*   Vpck  = (u32*)  (ws + OFF_VBF);
  u16*   attnb = (u16*)  (ws + OFF_ATTN);
  float* gate  = (float*)(ws + OFF_GATE);
  float* up    = (float*)(ws + OFF_UP);
  u16*   actb  = (u16*)  (ws + OFF_ACT);
  float* x2    = (float*)(ws + OFF_X2);
  u16*   h_bf  = (u16*)  (ws + OFF_HBF);

  auto gemm = [&](const u16* A, const float* B, float* C, int M, int N, int K) {
    if ((N & 127) == 0 && (M & 127) == 0) {
      dim3 g(N / 128, M / 128);
      gemm_bf16_fast<<<g, 256, 0, stream>>>(A, B, C, M, N, K);
    } else {
      dim3 g((N + 127) / 128, M / 64);
      gemm_bf16_gen<<<g, 256, 0, stream>>>(A, B, C, M, N, K);
    }
  };

  // 1) x = rmsnorm(hidden) -> bf16
  rmsnorm_bf16_kernel<<<SEQ, 256, 0, stream>>>(hidden, input_ln, x_bf, 2048, 2048, 2048);
  // 2) qa = x @ q_a_w
  gemm(x_bf, q_a_w, qa, SEQ, 1536, 2048);
  // 3) qin = rmsnorm(qa)
  rmsnorm_bf16_kernel<<<SEQ, 256, 0, stream>>>(qa, q_a_ln, qin, 1536, 1536, 1536);
  // 4) q = qin @ q_b_w
  gemm(qin, q_b_w, qf, SEQ, NHEAD * QHD_, 1536);
  // 5) ckv = x @ kv_a_w   (N=576 -> generic path)
  gemm(x_bf, kv_a_w, ckv, SEQ, 576, 2048);
  // 6) kvin = rmsnorm(ckv[:, :512])
  rmsnorm_bf16_kernel<<<SEQ, 256, 0, stream>>>(ckv, kv_a_ln, kvin, 512, 576, 512);
  // 7) kv = kvin @ kv_b_w
  gemm(kvin, kv_b_w, kvf, SEQ, 4096, 512);
  // 8-10) RoPE + head-major bf16 Q/K, pair-packed V
  {
    int tq = NHEAD * SEQ * QHD_;
    build_q_kernel<<<(tq + 255) / 256, 256, 0, stream>>>(qf, pos, cosb, sinb, Qbf);
    build_k_kernel<<<(tq + 255) / 256, 256, 0, stream>>>(kvf, ckv, pos, cosb, sinb, Kbf);
    int tv = NHEAD * (SEQ / 2) * VHD_;
    build_v_kernel<<<(tv + 255) / 256, 256, 0, stream>>>(kvf, Vpck);
  }
  // 11) flash attention
  mla_flash_attn<<<dim3(SEQ / 16, NHEAD), 32, 0, stream>>>(Qbf, Kbf, Vpck, attnb);
  // 12) o-proj into x2, 13) + residual
  gemm(attnb, o_w, x2, SEQ, 2048, 2048);
  {
    int n4 = SEQ * 2048 / 4;
    add_inplace_kernel<<<(n4 + 255) / 256, 256, 0, stream>>>(
        (float4*)x2, (const float4*)hidden, n4);
  }
  // 14) h = rmsnorm(x2)
  rmsnorm_bf16_kernel<<<SEQ, 256, 0, stream>>>(x2, post_ln, h_bf, 2048, 2048, 2048);
  // 15-17) MLP
  gemm(h_bf, gate_w, gate, SEQ, 8192, 2048);
  gemm(h_bf, up_w,   up,   SEQ, 8192, 2048);
  {
    int n2 = SEQ * 8192 / 2;
    silu_mul_kernel<<<(n2 + 255) / 256, 256, 0, stream>>>(
        (const float2*)gate, (const float2*)up, (u32*)actb, n2);
  }
  // 18) down-proj into out, 19) + residual
  gemm(actb, down_w, out, SEQ, 2048, 8192);
  {
    int n4 = SEQ * 2048 / 4;
    add_inplace_kernel<<<(n4 + 255) / 256, 256, 0, stream>>>(
        (float4*)out, (const float4*)x2, n4);
  }
}
